// EDGPAT_23785528885485
// MI455X (gfx1250) — compile-verified, matrix operand order checked by reading, not run on hardware
//
#include <hip/hip_runtime.h>
#include <hip/hip_bf16.h>

typedef __attribute__((ext_vector_type(16))) _Float16 v16h;
typedef __attribute__((ext_vector_type(8)))  float    v8f;

#define B_  4
#define F_  60082
#define C_  14695
#define D_  256
#define KN_ 64
#define KH_ 128
#define NH_ (B_ * KH_)   // 512 his rows

// Workspace layout (floats):
//  [0..255]    v   = W_proj @ w_ff
//  [256..383]  u   = W2 @ w_ff   (u[j] = W2[j,:].w_ff)
//  [384]       c0  = b_proj . w_ff
//  [385]       cb2 = b2 . w_ff
//  [386..389]  co2[b] = company_out[b] + b_fc + b_ff
//  [512..1023] his_dot[512]  (atomic accumulation of leaky-hidden . u)

// ---------------------------------------------------------------------------
// Kernel 1: tiny precompute (one block). Also zeroes the his_dot accumulator.
// ---------------------------------------------------------------------------
__global__ void edg_k1_precompute(const float* __restrict__ company_emb,
                                  const float* __restrict__ comp_table,
                                  const float* __restrict__ W_proj,
                                  const float* __restrict__ b_proj,
                                  const float* __restrict__ theta,
                                  const float* __restrict__ w_ff,
                                  const float* __restrict__ b_ff,
                                  const float* __restrict__ w_fc,
                                  const float* __restrict__ b_fc,
                                  const float* __restrict__ W2,
                                  const float* __restrict__ b2,
                                  const int*   __restrict__ com_id,
                                  float* __restrict__ ws) {
    const int t = threadIdx.x;           // 0..255
    // zero his_dot[512]
    ws[512 + t]       = 0.0f;
    ws[512 + 256 + t] = 0.0f;
    // v[t] = W_proj[t,:] . w_ff
    float acc = 0.0f;
    for (int d = 0; d < D_; ++d) acc += W_proj[t * D_ + d] * w_ff[d];
    ws[t] = acc;
    if (t < 128) {  // u[t] = W2[t,:] . w_ff
        float au = 0.0f;
        for (int d = 0; d < D_; ++d) au += W2[t * D_ + d] * w_ff[d];
        ws[256 + t] = au;
    }
    if (t == 0) {
        float c0 = 0.0f, cb2 = 0.0f;
        for (int d = 0; d < D_; ++d) { c0 += b_proj[d] * w_ff[d]; cb2 += b2[d] * w_ff[d]; }
        ws[384] = c0;
        ws[385] = cb2;
    }
    if (t < B_) {   // company_out[b] + b_fc + b_ff
        const int   cid = com_id[t];
        const float th  = theta[cid];
        float co = 0.0f;
        for (int d = 0; d < D_; ++d) {
            float cm = (1.0f - th) * company_emb[t * D_ + d] + th * comp_table[cid * D_ + d];
            co += cm * w_fc[d];
        }
        ws[386 + t] = co + b_fc[0] + b_ff[0];
    }
}

// ---------------------------------------------------------------------------
// Kernel 2: dense base output — streams field_table once (the HBM roofline).
// One wave per field row f:  sp = field_table[f].v + c0 ;
// out[b,f] = sp + co2[b] for all b.  Fully coalesced float4 loads.
// ---------------------------------------------------------------------------
__global__ void edg_k2_base(const float* __restrict__ field_table,
                            const float* __restrict__ ws,
                            float* __restrict__ out) {
    const int wave = threadIdx.x >> 5;
    const int lane = threadIdx.x & 31;
    const int f    = blockIdx.x * 8 + wave;
    if (f >= F_) return;
    const float4* ft = (const float4*)(field_table + (size_t)f * D_);
    const float4* vv = (const float4*)(ws);
    const float4  a0 = ft[lane * 2];
    const float4  a1 = ft[lane * 2 + 1];
    const float4  v0 = vv[lane * 2];
    const float4  v1 = vv[lane * 2 + 1];
    float acc = a0.x * v0.x + a0.y * v0.y + a0.z * v0.z + a0.w * v0.w
              + a1.x * v1.x + a1.y * v1.y + a1.z * v1.z + a1.w * v1.w;
    #pragma unroll
    for (int m = 16; m >= 1; m >>= 1) acc += __shfl_xor(acc, m, 32);
    if (lane == 0) {
        const float sp = acc + ws[384];   // + b_proj.w_ff
        #pragma unroll
        for (int b = 0; b < B_; ++b)
            out[(size_t)b * F_ + f] = sp + ws[386 + b];
    }
}

// ---------------------------------------------------------------------------
// Kernel 3: his-branch MLP hidden layer via WMMA f16.
// X (512 x 256) = raw_field_embed[his_nodes_flat],  W1 (256 x 128).
// Wave = one 16x16 tile of hidden; 8 fully-unrolled K-steps of 32 ->
// 8x v_wmma_f32_16x16x32_f16 with loads/cvt overlapped across WMMAs.
// Epilogue: leaky(h + b1) . u reduced over N-lanes, atomicAdd into his_dot.
// ---------------------------------------------------------------------------
__global__ void edg_k3_wmma_his(const float* __restrict__ raw_field_embed,
                                const float* __restrict__ W1,
                                const float* __restrict__ b1,
                                const int*   __restrict__ his_nodes,
                                float* __restrict__ ws) {
    const int wave  = threadIdx.x >> 5;    // ntile 0..7  (128 cols / 16)
    const int lane  = threadIdx.x & 31;
    const int mtile = blockIdx.x;          // 0..31 (512 rows / 16)
    const int m     = lane & 15;
    const int hi    = lane >> 4;           // 0 or 1
    const int kb    = hi * 8;              // K sub-offset per ISA A/B layout

    const int    row  = mtile * 16 + m;            // A-row this lane supplies
    const int    node = his_nodes[row];            // his_nodes flat (B*KH)
    const float* xr   = raw_field_embed + (size_t)node * D_;
    const int    col  = wave * 16 + m;             // B-col this lane supplies

    v8f c = {};
    #pragma unroll
    for (int kk = 0; kk < D_; kk += 32) {
        // ---- A fragment: 16x32 f16 (lane<16: K {0..7,16..23}; lane>=16: {8..15,24..31})
        const float4* xp = (const float4*)(xr + kk);
        const float4 a0 = xp[kb / 4];
        const float4 a1 = xp[kb / 4 + 1];
        const float4 a2 = xp[kb / 4 + 4];
        const float4 a3 = xp[kb / 4 + 5];
        v16h A;
        A[0]  = (_Float16)a0.x; A[1]  = (_Float16)a0.y; A[2]  = (_Float16)a0.z; A[3]  = (_Float16)a0.w;
        A[4]  = (_Float16)a1.x; A[5]  = (_Float16)a1.y; A[6]  = (_Float16)a1.z; A[7]  = (_Float16)a1.w;
        A[8]  = (_Float16)a2.x; A[9]  = (_Float16)a2.y; A[10] = (_Float16)a2.z; A[11] = (_Float16)a2.w;
        A[12] = (_Float16)a3.x; A[13] = (_Float16)a3.y; A[14] = (_Float16)a3.z; A[15] = (_Float16)a3.w;
        // ---- B fragment: 32x16 f16, symmetric K packing over column `col`
        v16h Bv;
        #pragma unroll
        for (int e = 0; e < 8; ++e) {
            Bv[e]     = (_Float16)W1[(kk + kb + e) * 128 + col];
            Bv[8 + e] = (_Float16)W1[(kk + 16 + kb + e) * 128 + col];
        }
        c = __builtin_amdgcn_wmma_f32_16x16x32_f16(false, A, false, Bv,
                                                   (short)0, c, false, false);
    }

    // Epilogue: D layout => lane holds (M = r + 8*hi, N = col) in c[r].
    const float b1c = b1[col];
    const float uc  = ws[256 + col];
    float part[8];
    #pragma unroll
    for (int r = 0; r < 8; ++r) {
        float h = c[r] + b1c;
        h = (h >= 0.0f) ? h : 0.01f * h;
        part[r] = h * uc;
    }
    #pragma unroll
    for (int r = 0; r < 8; ++r) {   // reduce over 16 N-lanes (masks < 16 stay in-group)
        float p = part[r];
        p += __shfl_xor(p, 1, 32);
        p += __shfl_xor(p, 2, 32);
        p += __shfl_xor(p, 4, 32);
        p += __shfl_xor(p, 8, 32);
        part[r] = p;
    }
    if (m == 0) {                    // lanes 0 and 16
        const int rbase = mtile * 16 + hi * 8;
        #pragma unroll
        for (int r = 0; r < 8; ++r)
            atomicAdd(&ws[512 + rbase + r], part[r]);
    }
}

// ---------------------------------------------------------------------------
// Kernel 4: scatter fixup with JAX .set semantics. Plain stores of the final
// value per touched (b,f): idempotent for duplicates; his-threads own the
// combined now+his case, now-threads defer if f also appears in his row.
// ---------------------------------------------------------------------------
__global__ void edg_k4_scatter(const float* __restrict__ field_emb,
                               const float* __restrict__ field_table,
                               const float* __restrict__ alpha_fields,
                               const float* __restrict__ w_ff,
                               const int*   __restrict__ now_nodes,
                               const int*   __restrict__ his_nodes,
                               const float* __restrict__ ws,
                               float* __restrict__ out) {
    const int t = blockIdx.x * blockDim.x + threadIdx.x;
    const int total_now = B_ * KN_;          // 256
    const int total     = total_now + NH_;   // 768
    if (t >= total) return;

    int b, f;
    float extra;
    if (t < total_now) {
        b = t / KN_;
        f = now_nodes[t];
        const int* hr = his_nodes + b * KH_;
        for (int j = 0; j < KH_; ++j)
            if (hr[j] == f) return;          // his-thread writes combined value
        float fe = 0.0f;
        for (int d = 0; d < D_; ++d) fe += field_emb[(size_t)f * D_ + d] * w_ff[d];
        extra = fe;
    } else {
        const int t2 = t - total_now;        // his position index 0..511
        b = t2 / KH_;
        f = his_nodes[t2];
        extra = ws[512 + t2] + ws[385];      // his_dot + b2.w_ff
        const int* nr = now_nodes + b * KN_;
        bool in_now = false;
        for (int j = 0; j < KN_; ++j)
            if (nr[j] == f) { in_now = true; break; }
        if (in_now) {
            float fe = 0.0f;
            for (int d = 0; d < D_; ++d) fe += field_emb[(size_t)f * D_ + d] * w_ff[d];
            extra += fe;
        }
    }
    const float alpha = alpha_fields[f];
    float sp = ws[384];
    for (int d = 0; d < D_; ++d) sp += field_table[(size_t)f * D_ + d] * ws[d];
    out[(size_t)b * F_ + f] = (1.0f - alpha) * sp + alpha * extra + ws[386 + b];
}

// ---------------------------------------------------------------------------
extern "C" void kernel_launch(void* const* d_in, const int* in_sizes, int n_in,
                              void* d_out, int out_size, void* d_ws, size_t ws_size,
                              hipStream_t stream) {
    const float* company_emb     = (const float*)d_in[0];
    const float* field_emb       = (const float*)d_in[1];
    const float* raw_field_embed = (const float*)d_in[2];
    const float* comp_table      = (const float*)d_in[3];
    const float* field_table     = (const float*)d_in[4];
    const float* W_proj          = (const float*)d_in[5];
    const float* b_proj          = (const float*)d_in[6];
    const float* theta           = (const float*)d_in[7];
    const float* alpha_fields    = (const float*)d_in[8];
    const float* w_ff            = (const float*)d_in[9];
    const float* b_ff            = (const float*)d_in[10];
    const float* w_fc            = (const float*)d_in[11];
    const float* b_fc            = (const float*)d_in[12];
    const float* W1              = (const float*)d_in[13];
    const float* b1              = (const float*)d_in[14];
    const float* W2              = (const float*)d_in[15];
    const float* b2              = (const float*)d_in[16];
    const int*   now_nodes       = (const int*)d_in[17];
    const int*   his_nodes       = (const int*)d_in[18];
    const int*   com_id          = (const int*)d_in[19];
    float* out = (float*)d_out;
    float* ws  = (float*)d_ws;

    edg_k1_precompute<<<1, 256, 0, stream>>>(company_emb, comp_table, W_proj, b_proj,
                                             theta, w_ff, b_ff, w_fc, b_fc, W2, b2,
                                             com_id, ws);
    edg_k2_base<<<(F_ + 7) / 8, 256, 0, stream>>>(field_table, ws, out);
    edg_k3_wmma_his<<<NH_ / 16, 256, 0, stream>>>(raw_field_embed, W1, b1, his_nodes, ws);
    edg_k4_scatter<<<3, 256, 0, stream>>>(field_emb, field_table, alpha_fields, w_ff,
                                          now_nodes, his_nodes, ws, out);
}
